// Method_GCN_11098195493080
// MI455X (gfx1250) — matrix operand order repaired and checked by reading, not compiled, not run on hardware
//
#include <hip/hip_runtime.h>
#include <hip/hip_bf16.h>
#include <stdint.h>

// ---- WMMA vector types (probe-verified signatures for gfx1250) ----
typedef __attribute__((ext_vector_type(16))) __bf16 v16bf;
typedef __attribute__((ext_vector_type(8)))  float  v8f;

// ============================================================================
// GEMM: Out[N, NOUT] = A[N, K] @ W[K, NOUT]   (bf16 WMMA, f32 accumulate)
//
// Block = WPB waves (wave32). The FULL weight matrix is staged once per block
// into LDS as bf16, transposed to [c][k] so each lane's B fragment
// (K = half*16 .. half*16+15, contiguous k order, same as A) is one contiguous 32B read.
// Each wave owns a 16-row tile and keeps NOUT/16 accumulators, so one
// A-fragment conversion feeds NOUT/16 WMMAs.
// ============================================================================
template<int K, int NOUT, int WPB>
__global__ __launch_bounds__(32 * WPB)
void wmma_gemm_bf16(const float* __restrict__ A, const float* __restrict__ W,
                    float* __restrict__ Out, int nrows) {
    constexpr int NT = NOUT / 16;          // N-tiles per wave
    __shared__ __bf16 lw[NOUT * K];        // transposed [c][k]

    const int tid = threadIdx.x;

    // Stage + convert full W (f32 global, coalesced (k,c) order -> bf16 LDS).
    for (int i = tid; i < NOUT * K; i += 32 * WPB) {
        const int k = i / NOUT, c = i % NOUT;       // W[i] == W[k*NOUT + c]
        lw[c * K + k] = (__bf16)W[i];
    }
    __syncthreads();

    const int lane = tid & 31;
    const int wave = tid >> 5;
    const int r    = lane & 15;            // column within tile / A row in tile
    const int half = lane >> 4;            // lane group 0/1
    const int tilebase = (blockIdx.x * WPB + wave) * 16;

    int row = tilebase + r;
    if (row >= nrows) row = nrows - 1;     // clamp: keeps EXEC all-1 for WMMA
    const float* __restrict__ arow = A + (size_t)row * K;

    v8f acc[NT];
    #pragma unroll
    for (int t = 0; t < NT; ++t) acc[t] = (v8f){};

    for (int k0 = 0; k0 < K; k0 += 32) {
        // ---- A fragment (contiguous, matches B):
        // elems 0..15 <- K = k0 + half*16 + 0..15
        const float4 f0 = *(const float4*)(arow + k0 + half * 16);
        const float4 f1 = *(const float4*)(arow + k0 + half * 16 + 4);
        const float4 f2 = *(const float4*)(arow + k0 + half * 16 + 8);
        const float4 f3 = *(const float4*)(arow + k0 + half * 16 + 12);
        v16bf a;
        a[0]  = (__bf16)f0.x; a[1]  = (__bf16)f0.y;
        a[2]  = (__bf16)f0.z; a[3]  = (__bf16)f0.w;
        a[4]  = (__bf16)f1.x; a[5]  = (__bf16)f1.y;
        a[6]  = (__bf16)f1.z; a[7]  = (__bf16)f1.w;
        a[8]  = (__bf16)f2.x; a[9]  = (__bf16)f2.y;
        a[10] = (__bf16)f2.z; a[11] = (__bf16)f2.w;
        a[12] = (__bf16)f3.x; a[13] = (__bf16)f3.y;
        a[14] = (__bf16)f3.z; a[15] = (__bf16)f3.w;

        // ---- NT B fragments from LDS; one WMMA each, A reused.
        #pragma unroll
        for (int t = 0; t < NT; ++t) {
            const v16bf b = *(const v16bf*)(lw + (t * 16 + r) * K + k0 + half * 16);
            acc[t] = __builtin_amdgcn_wmma_f32_16x16x32_bf16(
                /*neg_a=*/false, a, /*neg_b=*/false, b,
                /*c_mod=*/(short)0, acc[t], /*reuse_a=*/false, /*reuse_b=*/false);
        }
    }

    // C/D layout: element v -> row (8*half + v), col r. 16-lane coalesced.
    const int mrow = tilebase + half * 8;
    if (tilebase + 16 <= nrows) {          // uniform fast path: no per-row guard
        #pragma unroll
        for (int t = 0; t < NT; ++t)
            #pragma unroll
            for (int v = 0; v < 8; ++v)
                Out[(size_t)(mrow + v) * NOUT + t * 16 + r] = acc[t][v];
    } else if (tilebase < nrows) {         // tail tile only
        for (int t = 0; t < NT; ++t)
            for (int v = 0; v < 8; ++v)
                if (mrow + v < nrows)
                    Out[(size_t)(mrow + v) * NOUT + t * 16 + r] = acc[t][v];
    }
}

// ============================================================================
// Degree / normalization
// ============================================================================
__global__ void deg_init(float* deg, int n) {
    int i = blockIdx.x * blockDim.x + threadIdx.x;
    if (i < n) deg[i] = 1.0f;  // self loop
}
__global__ void deg_accum(const long long* __restrict__ dst, float* deg, long long e) {
    long long i = (long long)blockIdx.x * blockDim.x + threadIdx.x;
    if (i < e) atomicAdd(&deg[(int)dst[i]], 1.0f);
}
__global__ void deg_to_dinv(float* deg, int n) {
    int i = blockIdx.x * blockDim.x + threadIdx.x;
    if (i < n) deg[i] = rsqrtf(deg[i]);  // deg >= 1 always (self loops)
}

// ============================================================================
// Aggregation: out[d] = sum_e xw[src]*dinv[src]*dinv[dst] + self-loop + bias
// ============================================================================
template<int F>
__global__ void agg_init(const float* __restrict__ xw, const float* __restrict__ dinv,
                         const float* __restrict__ bias, float* __restrict__ out, int n) {
    constexpr int CH = F / 4;
    size_t idx = (size_t)blockIdx.x * blockDim.x + threadIdx.x;
    if (idx >= (size_t)n * CH) return;
    const int row = (int)(idx / CH);
    const int c   = (int)(idx % CH);
    const float di = dinv[row];
    const float di2 = di * di;                       // self-loop norm = dinv^2
    const float4 xv = *(const float4*)(xw + idx * 4);
    const float4 bv = *(const float4*)(bias + c * 4);
    float4 ov;
    ov.x = xv.x * di2 + bv.x; ov.y = xv.y * di2 + bv.y;
    ov.z = xv.z * di2 + bv.z; ov.w = xv.w * di2 + bv.w;
    *(float4*)(out + idx * 4) = ov;
}

template<int F>
__global__ void agg_edges(const float* __restrict__ xw, const float* __restrict__ dinv,
                          const long long* __restrict__ src, const long long* __restrict__ dst,
                          float* __restrict__ out, long long e) {
    constexpr int CH = F / 4;
    long long tid = (long long)blockIdx.x * blockDim.x + threadIdx.x;
    long long ei = tid / CH;
    if (ei >= e) return;
    const int c = (int)(tid % CH);
    const int s = (int)src[ei];
    const int d = (int)dst[ei];
    const float w = dinv[s] * dinv[d];
    const float4 xv = *(const float4*)(xw + (size_t)s * F + c * 4);
    float* op = out + (size_t)d * F + c * 4;
    atomicAdd(op + 0, xv.x * w);
    atomicAdd(op + 1, xv.y * w);
    atomicAdd(op + 2, xv.z * w);
    atomicAdd(op + 3, xv.w * w);
}

__global__ void relu4_kernel(float* h, size_t n4) {
    size_t i = (size_t)blockIdx.x * blockDim.x + threadIdx.x;
    if (i >= n4) return;
    float4 v = *(float4*)(h + i * 4);
    v.x = fmaxf(v.x, 0.0f); v.y = fmaxf(v.y, 0.0f);
    v.z = fmaxf(v.z, 0.0f); v.w = fmaxf(v.w, 0.0f);
    *(float4*)(h + i * 4) = v;
}

// log_softmax over 32 columns: one wave32 per row, lane = column.
__global__ void log_softmax32(float* __restrict__ o, int n) {
    const int row  = blockIdx.x * (blockDim.x >> 5) + (threadIdx.x >> 5);
    const int lane = threadIdx.x & 31;
    if (row >= n) return;
    const float v = o[(size_t)row * 32 + lane];
    float m = v;
    #pragma unroll
    for (int s = 16; s > 0; s >>= 1) m = fmaxf(m, __shfl_xor(m, s, 32));
    float sum = __expf(v - m);
    #pragma unroll
    for (int s = 16; s > 0; s >>= 1) sum += __shfl_xor(sum, s, 32);
    o[(size_t)row * 32 + lane] = v - m - __logf(sum);
}

// ============================================================================
extern "C" void kernel_launch(void* const* d_in, const int* in_sizes, int n_in,
                              void* d_out, int out_size, void* d_ws, size_t ws_size,
                              hipStream_t stream) {
    const float*     x   = (const float*)d_in[0];
    const long long* ei  = (const long long*)d_in[1];   // int64 [2, E]
    const float*     W1  = (const float*)d_in[2];
    const float*     b1  = (const float*)d_in[3];
    const float*     W2  = (const float*)d_in[4];
    const float*     b2  = (const float*)d_in[5];
    float*           out = (float*)d_out;

    const int  IN  = 512, HID = 64, OUT = 32;
    const int  N   = in_sizes[0] / IN;
    const long long E = in_sizes[1] / 2;
    const long long* src = ei;
    const long long* dst = ei + E;

    // Workspace layout (256B aligned): dinv | XW (N*64, reused as N*32) | H (N*64)
    char* ws = (char*)d_ws;
    size_t off = 0;
    float* dinv = (float*)(ws + off); off += (((size_t)N * 4) + 255) & ~(size_t)255;
    float* xw   = (float*)(ws + off); off += (((size_t)N * HID * 4) + 255) & ~(size_t)255;
    float* h    = (float*)(ws + off);

    const int T = 256;
    constexpr int WPB = 8;                  // waves per GEMM block
    const int rows_per_block = 16 * WPB;
    const int gblocks = (N + rows_per_block - 1) / rows_per_block;

    // --- normalization: deg -> dinv (in place) ---
    deg_init<<<(N + T - 1) / T, T, 0, stream>>>(dinv, N);
    deg_accum<<<(unsigned)((E + T - 1) / T), T, 0, stream>>>(dst, dinv, E);
    deg_to_dinv<<<(N + T - 1) / T, T, 0, stream>>>(dinv, N);

    // --- layer 1: XW1 = x @ W1 (WMMA bf16) ---
    wmma_gemm_bf16<512, 64, WPB><<<gblocks, 32 * WPB, 0, stream>>>(x, W1, xw, N);

    // --- aggregate + bias, then relu -> H ---
    {
        size_t total4 = (size_t)N * (HID / 4);
        agg_init<64><<<(unsigned)((total4 + T - 1) / T), T, 0, stream>>>(xw, dinv, b1, h, N);
        long long work = E * (HID / 4);
        agg_edges<64><<<(unsigned)((work + T - 1) / T), T, 0, stream>>>(xw, dinv, src, dst, h, E);
        relu4_kernel<<<(unsigned)((total4 + T - 1) / T), T, 0, stream>>>(h, total4);
    }

    // --- layer 2: XW2 = H @ W2 (WMMA bf16), reuse xw buffer ---
    wmma_gemm_bf16<64, 32, WPB><<<gblocks, 32 * WPB, 0, stream>>>(h, W2, xw, N);

    // --- aggregate + bias straight into d_out ---
    {
        size_t total4 = (size_t)N * (OUT / 4);
        agg_init<32><<<(unsigned)((total4 + T - 1) / T), T, 0, stream>>>(xw, dinv, b2, out, N);
        long long work = E * (OUT / 4);
        agg_edges<32><<<(unsigned)((work + T - 1) / T), T, 0, stream>>>(xw, dinv, src, dst, out, E);
    }

    // --- log_softmax in place on d_out: one wave per row ---
    log_softmax32<<<(N * 32 + T - 1) / T, T, 0, stream>>>(out, N);
}
